// AttnDecoder_83597243449431
// MI455X (gfx1250) — matrix operand order repaired and checked by reading, compiled, not run on hardware
//
#include <hip/hip_runtime.h>
#include <hip/hip_bf16.h>

typedef __attribute__((ext_vector_type(16))) _Float16 v16h;
typedef __attribute__((ext_vector_type(8)))  float    v8f;

#define B_ 256
#define T_ 64
#define C_ 1024
#define H_ 1024

// ---------- WMMA fragment packing (CDNA5 ISA 7.12.2 layouts) ----------
// A tile 16x32 f16: lane = (m%16) + hi*16 ; per lane 16 halves:
//   j<8  -> K = k0 + hi*8 + j
//   j>=8 -> K = k0 + 16 + hi*8 + (j-8)
static __device__ __forceinline__ size_t packA_index(int m, int k, int K) {
    int mt = m >> 4, r = m & 15;
    int kt = k >> 5, kin = k & 31;
    int hi = (kin >> 3) & 1;
    int j  = (kin & 7) + ((kin & 16) ? 8 : 0);
    int lane = r + (hi << 4);
    return ((size_t)mt * (size_t)(K >> 5) + (size_t)kt) * 512 + (size_t)lane * 16 + j;
}
// B tile 32x16 f16: lane = (n%16) + hi*16 ; per lane halves j<8 -> K = k0 + hi*8 + j, j>=8 -> K = k0 + 16 + hi*8 + (j-8)
static __device__ __forceinline__ size_t packB_index(int k, int n, int K) {
    int nt = n >> 4, c = n & 15;
    int kt = k >> 5, kin = k & 31;
    int hi = (kin >> 3) & 1;
    int j  = (kin & 7) + ((kin & 16) ? 8 : 0);
    int lane = c + (hi << 4);
    return ((size_t)nt * (size_t)(K >> 5) + (size_t)kt) * 512 + (size_t)lane * 16 + j;
}

static __device__ __forceinline__ float sigm(float x) { return 1.0f / (1.0f + expf(-x)); }

// ---------- prep kernels ----------
__global__ void k_pack_h(const float* __restrict__ h, _Float16* __restrict__ dst) {
    size_t gid = (size_t)blockIdx.x * blockDim.x + threadIdx.x;   // B*T*C threads
    int m = (int)(gid / C_), k = (int)(gid % C_);
    dst[packA_index(m, k, C_)] = (_Float16)h[gid];
}

// src row-major [N][K]; math B-matrix is B[k][n] = src[n][k]
__global__ void k_packB(const float* __restrict__ src, _Float16* __restrict__ dst, int N, int K) {
    size_t gid = (size_t)blockIdx.x * blockDim.x + threadIdx.x;   // N*K threads
    int n = (int)(gid / K), k = (int)(gid % K);
    dst[packB_index(k, n, K)] = (_Float16)src[gid];
}

__global__ void k_zero(float* __restrict__ p) {
    size_t gid = (size_t)blockIdx.x * blockDim.x + threadIdx.x;
    p[gid] = 0.0f;
}

// ---------- WMMA GEMM, 1x4 tile blocking ----------
// Each wave computes a 16x64 strip of C: A fragment loaded once per k-step,
// fed to 4 independent WMMA accumulator chains (latency hiding + A reuse).
__global__ __launch_bounds__(256) void k_gemm4(const _Float16* __restrict__ Apk,
                                               const _Float16* __restrict__ Bpk,
                                               const float* __restrict__ bias,
                                               float* __restrict__ Cf,
                                               _Float16* __restrict__ Ch,
                                               int N, int K) {
    int grp = blockIdx.x * (blockDim.x >> 5) + (threadIdx.x >> 5);
    int Ng = N >> 6;                       // 4-tile groups along N
    int mt = grp / Ng, ng = grp % Ng;
    int nt0 = ng << 2;
    int lane = threadIdx.x & 31;
    int ktn = K >> 5;
    const v16h* ap  = (const v16h*)Apk + (size_t)mt * ktn * 32 + lane;
    const v16h* bp0 = (const v16h*)Bpk + (size_t)(nt0 + 0) * ktn * 32 + lane;
    const v16h* bp1 = (const v16h*)Bpk + (size_t)(nt0 + 1) * ktn * 32 + lane;
    const v16h* bp2 = (const v16h*)Bpk + (size_t)(nt0 + 2) * ktn * 32 + lane;
    const v16h* bp3 = (const v16h*)Bpk + (size_t)(nt0 + 3) * ktn * 32 + lane;
    v8f acc0 = {}, acc1 = {}, acc2 = {}, acc3 = {};
#pragma unroll 2
    for (int kt = 0; kt < ktn; ++kt) {
        size_t o = (size_t)kt * 32;
        v16h a  = ap[o];
        v16h b0 = bp0[o];
        v16h b1 = bp1[o];
        v16h b2 = bp2[o];
        v16h b3 = bp3[o];
        acc0 = __builtin_amdgcn_wmma_f32_16x16x32_f16(false, a, false, b0, (short)0, acc0, false, false);
        acc1 = __builtin_amdgcn_wmma_f32_16x16x32_f16(false, a, false, b1, (short)0, acc1, false, false);
        acc2 = __builtin_amdgcn_wmma_f32_16x16x32_f16(false, a, false, b2, (short)0, acc2, false, false);
        acc3 = __builtin_amdgcn_wmma_f32_16x16x32_f16(false, a, false, b3, (short)0, acc3, false, false);
    }
    int m0 = (mt << 4) + ((lane >> 4) << 3);
    v8f accs[4] = {acc0, acc1, acc2, acc3};
#pragma unroll
    for (int i = 0; i < 4; ++i) {
        int n = ((nt0 + i) << 4) + (lane & 15);
        float bv = bias ? bias[n] : 0.0f;
        if (Ch) {
#pragma unroll
            for (int v = 0; v < 8; ++v) Ch[(size_t)(m0 + v) * N + n] = (_Float16)(accs[i][v] + bv);
        } else {
#pragma unroll
            for (int v = 0; v < 8; ++v) Cf[(size_t)(m0 + v) * N + n] = accs[i][v] + bv;
        }
    }
}

// ---------- per-step kernels ----------
// z3[b,t] = sum_c tanh(z1[b,c] + z2[b,t,c]) * Wa3[c] + ba3   (one wave per (b,t))
__global__ __launch_bounds__(256) void k_z3(const float* __restrict__ z1,
                                            const _Float16* __restrict__ z2h,
                                            const float* __restrict__ Wa3,
                                            const float* __restrict__ ba3,
                                            float* __restrict__ z3) {
    int bt = blockIdx.x * (blockDim.x >> 5) + (threadIdx.x >> 5);
    int lane = threadIdx.x & 31;
    int b = bt / T_;
    const float*    z1p = z1  + (size_t)b  * C_;
    const _Float16* z2p = z2h + (size_t)bt * C_;
    float acc = 0.0f;
    for (int c = lane; c < C_; c += 32)
        acc += tanhf(z1p[c] + (float)z2p[c]) * Wa3[c];
    for (int off = 16; off; off >>= 1) acc += __shfl_xor(acc, off);
    if (lane == 0) z3[bt] = acc + ba3[0];
}

// softmax over T=64 per batch row, in place (one wave per b, 2 elems/lane)
__global__ __launch_bounds__(256) void k_softmax(float* __restrict__ z3) {
    int b = blockIdx.x * (blockDim.x >> 5) + (threadIdx.x >> 5);
    int lane = threadIdx.x & 31;
    float* p = z3 + (size_t)b * T_;
    float v0 = p[lane], v1 = p[lane + 32];
    float mx = fmaxf(v0, v1);
    for (int off = 16; off; off >>= 1) mx = fmaxf(mx, __shfl_xor(mx, off));
    float e0 = expf(v0 - mx), e1 = expf(v1 - mx);
    float sm = e0 + e1;
    for (int off = 16; off; off >>= 1) sm += __shfl_xor(sm, off);
    float inv = 1.0f / sm;
    p[lane] = e0 * inv; p[lane + 32] = e1 * inv;
}

// ct[b,c] = sum_t beta[b,t] * h[b,t,c]
__global__ __launch_bounds__(256) void k_ct(const float* __restrict__ beta,
                                            const float* __restrict__ h,
                                            float* __restrict__ ct) {
    size_t gid = (size_t)blockIdx.x * blockDim.x + threadIdx.x;   // B*C
    int b = (int)(gid / C_), c = (int)(gid % C_);
    const float* hp = h + (size_t)b * T_ * C_ + c;
    const float* bp = beta + (size_t)b * T_;
    float acc = 0.0f;
#pragma unroll 8
    for (int t = 0; t < T_; ++t) acc += bp[t] * hp[(size_t)t * C_];
    ct[gid] = acc;
}

// y_tilde[b] = b_tilde + Wt[0]*y_seq[b,t] + sum_c Wt[1+c]*ct[b,c]
__global__ __launch_bounds__(256) void k_ytilde(const float* __restrict__ ct,
                                                const float* __restrict__ y_seq, int t,
                                                const float* __restrict__ Wt,
                                                const float* __restrict__ bt_,
                                                float* __restrict__ yt) {
    int b = blockIdx.x * (blockDim.x >> 5) + (threadIdx.x >> 5);
    int lane = threadIdx.x & 31;
    const float* cp = ct + (size_t)b * C_;
    float acc = 0.0f;
    for (int c = lane; c < C_; c += 32) acc += Wt[1 + c] * cp[c];
    for (int off = 16; off; off >>= 1) acc += __shfl_xor(acc, off);
    if (lane == 0) yt[b] = acc + Wt[0] * y_seq[(size_t)b * T_ + t] + bt_[0];
}

// LSTM gate nonlinearity + state update; writes next-step packed A fragments.
__global__ __launch_bounds__(256) void k_lstm(const float* __restrict__ gates,
                                              const float* __restrict__ yt,
                                              const float* __restrict__ W_ih,
                                              const float* __restrict__ b_ih,
                                              const float* __restrict__ b_hh,
                                              float* __restrict__ d, float* __restrict__ s,
                                              _Float16* __restrict__ x16,
                                              _Float16* __restrict__ d16) {
    size_t gid = (size_t)blockIdx.x * blockDim.x + threadIdx.x;   // B*H
    int b = (int)(gid / H_), j = (int)(gid % H_);
    const float* g = gates + (size_t)b * 4 * H_;
    float ytb = yt[b];
    float gi = g[j]          + b_ih[j]          + b_hh[j]          + ytb * W_ih[j];
    float gf = g[H_ + j]     + b_ih[H_ + j]     + b_hh[H_ + j]     + ytb * W_ih[H_ + j];
    float gg = g[2 * H_ + j] + b_ih[2 * H_ + j] + b_hh[2 * H_ + j] + ytb * W_ih[2 * H_ + j];
    float go = g[3 * H_ + j] + b_ih[3 * H_ + j] + b_hh[3 * H_ + j] + ytb * W_ih[3 * H_ + j];
    float sn = sigm(gf) * s[gid] + sigm(gi) * tanhf(gg);
    float dn = sigm(go) * tanhf(sn);
    s[gid] = sn; d[gid] = dn;
    x16[packA_index(b, j,      2 * H_)] = (_Float16)dn;
    x16[packA_index(b, H_ + j, 2 * H_)] = (_Float16)sn;
    d16[packA_index(b, j, H_)]          = (_Float16)dn;
}

// cat16 = packed A fragments of concat([d, ct], axis=1)  (K = H+C)
__global__ __launch_bounds__(256) void k_cat(const float* __restrict__ d,
                                             const float* __restrict__ ct,
                                             _Float16* __restrict__ cat16) {
    size_t gid = (size_t)blockIdx.x * blockDim.x + threadIdx.x;   // B*(H+C)
    const int K = H_ + C_;
    int b = (int)(gid / K), k = (int)(gid % K);
    float v = (k < H_) ? d[(size_t)b * H_ + k] : ct[(size_t)b * C_ + (k - H_)];
    cat16[packA_index(b, k, K)] = (_Float16)v;
}

// out[b] = b_fc2 + sum_h state[b,h] * W_fc2[h]
__global__ __launch_bounds__(256) void k_out(const float* __restrict__ state,
                                             const float* __restrict__ Wfc2,
                                             const float* __restrict__ bfc2,
                                             float* __restrict__ out) {
    int b = blockIdx.x * (blockDim.x >> 5) + (threadIdx.x >> 5);
    int lane = threadIdx.x & 31;
    const float* sp = state + (size_t)b * H_;
    float acc = 0.0f;
    for (int k = lane; k < H_; k += 32) acc += sp[k] * Wfc2[k];
    for (int off = 16; off; off >>= 1) acc += __shfl_xor(acc, off);
    if (lane == 0) out[b] = acc + bfc2[0];
}

extern "C" void kernel_launch(void* const* d_in, const int* in_sizes, int n_in,
                              void* d_out, int out_size, void* d_ws, size_t ws_size,
                              hipStream_t stream) {
    const float* h     = (const float*)d_in[0];
    const float* y_seq = (const float*)d_in[1];
    const float* W_a1  = (const float*)d_in[2];
    const float* b_a1  = (const float*)d_in[3];
    const float* W_a2  = (const float*)d_in[4];
    const float* b_a2  = (const float*)d_in[5];
    const float* W_a3  = (const float*)d_in[6];
    const float* b_a3  = (const float*)d_in[7];
    const float* W_ih  = (const float*)d_in[8];
    const float* W_hh  = (const float*)d_in[9];
    const float* b_ih  = (const float*)d_in[10];
    const float* b_hh  = (const float*)d_in[11];
    const float* W_t   = (const float*)d_in[12];
    const float* b_t   = (const float*)d_in[13];
    const float* W_fc1 = (const float*)d_in[14];
    const float* b_fc1 = (const float*)d_in[15];
    const float* W_fc2 = (const float*)d_in[16];
    const float* b_fc2 = (const float*)d_in[17];
    float* out = (float*)d_out;

    // ---- workspace carve-up (all sizes multiples of 256B) ----
    char* w = (char*)d_ws;
    size_t off = 0;
    auto carve = [&](size_t bytes) -> char* { char* p = w + off; off += (bytes + 255) & ~(size_t)255; return p; };
    _Float16* z2h    = (_Float16*)carve((size_t)B_ * T_ * C_ * 2);   // 32 MB, f16 z2
    _Float16* h16pk  = (_Float16*)carve((size_t)B_ * T_ * C_ * 2);   // 32 MB, packed A of h
    _Float16* W1Tpk  = (_Float16*)carve((size_t)2 * H_ * C_ * 2);    // 4 MB
    _Float16* W2Tpk  = (_Float16*)carve((size_t)C_ * C_ * 2);        // 2 MB
    _Float16* WhhPk  = (_Float16*)carve((size_t)H_ * 4 * H_ * 2);    // 8 MB
    _Float16* Wfc1Pk = (_Float16*)carve((size_t)(C_ + H_) * H_ * 2); // 4 MB
    // contiguous zero region: d, s, x16pk, d16pk
    float*    dbuf   = (float*)   carve((size_t)B_ * H_ * 4);
    float*    sbuf   = (float*)   carve((size_t)B_ * H_ * 4);
    _Float16* x16pk  = (_Float16*)carve((size_t)B_ * 2 * H_ * 2);
    _Float16* d16pk  = (_Float16*)carve((size_t)B_ * H_ * 2);
    float*    z1buf  = (float*)   carve((size_t)B_ * C_ * 4);
    float*    z3buf  = (float*)   carve((size_t)B_ * T_ * 4);
    float*    ctbuf  = (float*)   carve((size_t)B_ * C_ * 4);
    float*    ytil   = (float*)   carve((size_t)B_ * 4);
    float*    gates  = (float*)   carve((size_t)B_ * 4 * H_ * 4);
    _Float16* cat16  = (_Float16*)carve((size_t)B_ * (C_ + H_) * 2);
    float*    state  = (float*)   carve((size_t)B_ * H_ * 4);

    // ---- prep: pack operands, zero recurrent state ----
    k_pack_h<<<(B_ * T_ * C_) / 256, 256, 0, stream>>>(h, h16pk);
    k_packB<<<(C_ * 2 * H_) / 256, 256, 0, stream>>>(W_a1, W1Tpk, C_, 2 * H_);
    k_packB<<<(C_ * C_) / 256, 256, 0, stream>>>(W_a2, W2Tpk, C_, C_);
    k_packB<<<(4 * H_ * H_) / 256, 256, 0, stream>>>(W_hh, WhhPk, 4 * H_, H_);
    k_packB<<<(H_ * (C_ + H_)) / 256, 256, 0, stream>>>(W_fc1, Wfc1Pk, H_, C_ + H_);
    {   // zero d, s, x16pk, d16pk (3.5 MB contiguous)
        int nzero = (B_ * H_ * 4 * 2 + B_ * 2 * H_ * 2 + B_ * H_ * 2) / 4;
        k_zero<<<nzero / 256, 256, 0, stream>>>(dbuf);
    }
    // loop-invariant z2 = h @ W_attn2^T + b_attn2   (M=16384, N=1024, K=1024) -> f16
    k_gemm4<<<((B_ * T_ / 16) * (C_ / 64)) / 8, 256, 0, stream>>>(
        h16pk, W2Tpk, b_a2, nullptr, z2h, C_, C_);

    // ---- sequential scan: t = 0 .. T-2 ----
    for (int t = 0; t < T_ - 1; ++t) {
        // z1 = [d;s] @ W_attn1^T + b_attn1   (M=256, N=1024, K=2048)
        k_gemm4<<<((B_ / 16) * (C_ / 64)) / 8, 256, 0, stream>>>(
            x16pk, W1Tpk, b_a1, z1buf, nullptr, C_, 2 * H_);
        k_z3<<<(B_ * T_) / 8, 256, 0, stream>>>(z1buf, z2h, W_a3, b_a3, z3buf);
        k_softmax<<<B_ / 8, 256, 0, stream>>>(z3buf);
        k_ct<<<(B_ * C_) / 256, 256, 0, stream>>>(z3buf, h, ctbuf);
        k_ytilde<<<B_ / 8, 256, 0, stream>>>(ctbuf, y_seq, t, W_t, b_t, ytil);
        // gates_raw = d @ W_hh^T   (M=256, N=4096, K=1024)
        k_gemm4<<<((B_ / 16) * (4 * H_ / 64)) / 8, 256, 0, stream>>>(
            d16pk, WhhPk, nullptr, gates, nullptr, 4 * H_, H_);
        k_lstm<<<(B_ * H_) / 256, 256, 0, stream>>>(
            gates, ytil, W_ih, b_ih, b_hh, dbuf, sbuf, x16pk, d16pk);
    }

    // ---- final attention + head ----
    k_gemm4<<<((B_ / 16) * (C_ / 64)) / 8, 256, 0, stream>>>(
        x16pk, W1Tpk, b_a1, z1buf, nullptr, C_, 2 * H_);
    k_z3<<<(B_ * T_) / 8, 256, 0, stream>>>(z1buf, z2h, W_a3, b_a3, z3buf);
    k_softmax<<<B_ / 8, 256, 0, stream>>>(z3buf);
    k_ct<<<(B_ * C_) / 256, 256, 0, stream>>>(z3buf, h, ctbuf);
    k_cat<<<(B_ * (C_ + H_)) / 256, 256, 0, stream>>>(dbuf, ctbuf, cat16);
    // state = concat(d, ct) @ W_fc1^T + b_fc1   (M=256, N=1024, K=2048)
    k_gemm4<<<((B_ / 16) * (H_ / 64)) / 8, 256, 0, stream>>>(
        cat16, Wfc1Pk, b_fc1, state, nullptr, H_, C_ + H_);
    k_out<<<B_ / 8, 256, 0, stream>>>(state, W_fc2, b_fc2, out);
}
